// DeformableEncoderBlock_9371618640465
// MI455X (gfx1250) — matrix operand order repaired and checked by reading, not compiled, not run on hardware
//
#include <hip/hip_runtime.h>
#include <math.h>

// ---------------- constants (match reference) ----------------
#define EMB      256
#define N_HEADS  8
#define N_LVLS   4
#define N_POINTS 4
#define HEAD_DIM 32
#define D_FFN    1024
#define LEN_TOK  13294   // 100*100 + 50*50 + 25*25 + 13*13
#define BATCH    4
#define NTOK     (BATCH * LEN_TOK)   // 53176

typedef __attribute__((ext_vector_type(8)))  __bf16 v8bf;
typedef __attribute__((ext_vector_type(16))) __bf16 v16bf;
typedef __attribute__((ext_vector_type(8)))  float  v8f;

// =====================================================================
// Weight pre-pack: fp32 [K,N] row-major -> bf16 WMMA-B fragment order.
// Tile (kt,nt) covers K rows [kt*32,kt*32+32), N cols [nt*16,nt*16+16).
// Within a tile, element index = lane*16 + e with
//   k = kt*32 + e + 16*(lane>=16),  n = nt*16 + (lane&15)
// so each lane's 16 B-fragment elements are contiguous (32B load).
// =====================================================================
__global__ void pack_b_kernel(const float* __restrict__ W, __bf16* __restrict__ out,
                              int K, int N) {
    size_t gid = (size_t)blockIdx.x * blockDim.x + threadIdx.x;
    size_t total = (size_t)K * N;
    if (gid >= total) return;
    int ntiles = N >> 4;
    size_t tile = gid >> 9;          // /512 elements per tile
    int rem = (int)(gid & 511);
    int l = rem >> 4;
    int e = rem & 15;
    int kt = (int)(tile / ntiles);
    int nt = (int)(tile % ntiles);
    int k = kt * 32 + e + ((l >> 4) << 4);
    int n = nt * 16 + (l & 15);
    out[gid] = (__bf16)W[(size_t)k * N + n];
}

// q = bf16(features + pos); feat_bf16 = bf16(features)
__global__ void prep_kernel(const float* __restrict__ feat, const float* __restrict__ pos,
                            __bf16* __restrict__ feat_bf, __bf16* __restrict__ q_bf,
                            size_t n) {
    size_t i = (size_t)blockIdx.x * blockDim.x + threadIdx.x;
    if (i >= n) return;
    float f = feat[i];
    feat_bf[i] = (__bf16)f;
    q_bf[i] = (__bf16)(f + pos[i]);
}

__device__ __forceinline__ void store_tile(void* Cout, const v8f& acc, float bv,
                                           int m0, int hi, int n, int N, int M,
                                           bool relu_bf16) {
#pragma unroll
    for (int e = 0; e < 8; ++e) {
        int row = m0 + e + hi * 8;            // C/D layout: row by VGPR index + half
        if (row < M) {
            float v = acc[e] + bv;
            if (relu_bf16) {
                v = v > 0.f ? v : 0.f;
                ((__bf16*)Cout)[(size_t)row * N + n] = (__bf16)v;
            } else {
                ((float*)Cout)[(size_t)row * N + n] = v;
            }
        }
    }
}

// =====================================================================
// K=256 GEMM with the A row panel staged in LDS (per-wave 8KB region,
// stored directly in fragment order). A read from global exactly once;
// per N-panel the fragments are re-read via ds_load_b128 (in-order,
// wave-private -> no barrier). Only 1 A-frag + 8 accs live in VGPRs.
// block = 256 thr = 8 waves; wave -> 16 rows x N cols. grid = (ceil(M/128),1)
// =====================================================================
template <int NPANELS, bool RELU_BF16_OUT>
__global__ __launch_bounds__(256, 1) void gemm_wmma_kc_kernel(
        const __bf16* __restrict__ A,
        const __bf16* __restrict__ Bp,
        const float* __restrict__ bias,
        void* __restrict__ Cout,
        int M, int N) {
    __shared__ __align__(16) __bf16 ldsA[8 * 8 * 512];   // 8 waves x 8 frags x 512 bf16 = 64KB

    const int wave = threadIdx.x >> 5;
    const int lane = threadIdx.x & 31;
    const int m0   = (blockIdx.x * 8 + wave) * 16;
    if (m0 >= M) return;                       // wave-uniform; LDS regions are wave-private
    const int l15  = lane & 15;
    const int hi   = lane >> 4;
    const int ntiles_total = N >> 4;

    int m = m0 + l15;
    int mc = (m < M) ? m : (M - 1);            // clamp loads; stores are guarded
    const __bf16* Arow = A + (size_t)mc * 256;
    const int kb = hi ? 16 : 0;                // contiguous k = 16*hi + e (matches packed B)

    __bf16* myA = &ldsA[wave * 4096];
#pragma unroll
    for (int kt8 = 0; kt8 < 8; ++kt8) {
        v8bf alo = *reinterpret_cast<const v8bf*>(Arow + kt8 * 32 + kb);
        v8bf ahi = *reinterpret_cast<const v8bf*>(Arow + kt8 * 32 + kb + 8);
        *reinterpret_cast<v8bf*>(&myA[(kt8 * 32 + lane) * 16])     = alo;
        *reinterpret_cast<v8bf*>(&myA[(kt8 * 32 + lane) * 16 + 8]) = ahi;
    }

#pragma unroll 1                               // one panel's accs live at a time
    for (int np = 0; np < NPANELS; ++np) {
        const int n0 = np * 128;
        v8f acc[8];
#pragma unroll
        for (int j = 0; j < 8; ++j) acc[j] = (v8f){};

#pragma unroll
        for (int kt8 = 0; kt8 < 8; ++kt8) {
            v16bf afrag = *reinterpret_cast<const v16bf*>(&myA[(kt8 * 32 + lane) * 16]);
            size_t tbase = ((size_t)kt8 * ntiles_total + (n0 >> 4)) * 512 + (size_t)lane * 16;
#pragma unroll
            for (int j = 0; j < 8; ++j) {
                v16bf bf = *reinterpret_cast<const v16bf*>(Bp + tbase + (size_t)j * 512);
                acc[j] = __builtin_amdgcn_wmma_f32_16x16x32_bf16(false, afrag, false, bf,
                                                                 (short)0, acc[j], false, false);
            }
        }
#pragma unroll
        for (int j = 0; j < 8; ++j) {
            int n = n0 + j * 16 + l15;
            store_tile(Cout, acc[j], bias[n], m0, hi, n, N, M, RELU_BF16_OUT);
        }
    }
}

// =====================================================================
// Streaming GEMM (any K multiple of 32): K-outer, 8 N-tiles resident.
// grid = (ceil(M/128), N/128). Used for FFN2 (K=1024).
// =====================================================================
template <bool RELU_BF16_OUT>
__global__ __launch_bounds__(256, 1) void gemm_wmma_stream_kernel(
        const __bf16* __restrict__ A,
        const __bf16* __restrict__ Bp,
        const float* __restrict__ bias,
        void* __restrict__ Cout,
        int M, int N, int K) {
    const int wave = threadIdx.x >> 5;
    const int lane = threadIdx.x & 31;
    const int m0   = (blockIdx.x * 8 + wave) * 16;
    if (m0 >= M) return;
    const int n0   = blockIdx.y * 128;
    const int l15  = lane & 15;
    const int hi   = lane >> 4;
    const int ntiles_total = N >> 4;

    int m = m0 + l15;
    int mc = (m < M) ? m : (M - 1);
    const __bf16* Arow = A + (size_t)mc * K;
    const int kb = hi ? 16 : 0;

    v8f acc[8];
#pragma unroll
    for (int j = 0; j < 8; ++j) acc[j] = (v8f){};

    for (int kt = 0; kt < K; kt += 32) {
        if (kt + 32 < K) __builtin_prefetch(Arow + kt + 32 + kb, 0, 0);  // global_prefetch_b8

        v8bf alo = *reinterpret_cast<const v8bf*>(Arow + kt + kb);
        v8bf ahi = *reinterpret_cast<const v8bf*>(Arow + kt + kb + 8);
        v16bf afrag;
#pragma unroll
        for (int i = 0; i < 8; ++i) { afrag[i] = alo[i]; afrag[i + 8] = ahi[i]; }

        size_t tbase = ((size_t)(kt >> 5) * ntiles_total + (n0 >> 4)) * 512 + (size_t)lane * 16;
#pragma unroll
        for (int j = 0; j < 8; ++j) {
            v16bf bf = *reinterpret_cast<const v16bf*>(Bp + tbase + (size_t)j * 512);
            acc[j] = __builtin_amdgcn_wmma_f32_16x16x32_bf16(false, afrag, false, bf,
                                                             (short)0, acc[j], false, false);
        }
    }

#pragma unroll
    for (int j = 0; j < 8; ++j) {
        int n = n0 + j * 16 + l15;
        store_tile(Cout, acc[j], bias[n], m0, hi, n, N, M, RELU_BF16_OUT);
    }
}

// softmax over 16 contiguous values per (token, head)
__global__ void softmax16_kernel(float* __restrict__ attn, int ngroups) {
    int g = blockIdx.x * blockDim.x + threadIdx.x;
    if (g >= ngroups) return;
    float* p = attn + (size_t)g * 16;
    float mx = p[0];
#pragma unroll
    for (int i = 1; i < 16; ++i) mx = fmaxf(mx, p[i]);
    float s = 0.f;
    float e[16];
#pragma unroll
    for (int i = 0; i < 16; ++i) { e[i] = __expf(p[i] - mx); s += e[i]; }
    float inv = 1.f / s;
#pragma unroll
    for (int i = 0; i < 16; ++i) p[i] = e[i] * inv;
}

// =====================================================================
// Deformable sampling: block = 1 token (256 thr), wave = head,
// lane = head-dim channel -> corner gathers are 128B contiguous.
// Writes bf16 directly (input to W_out GEMM).
// =====================================================================
__global__ void deform_sample_kernel(const float* __restrict__ value,
                                     const float* __restrict__ off,
                                     const float* __restrict__ attn,
                                     const float* __restrict__ refp,
                                     __bf16* __restrict__ samp_out) {
    const int tok = blockIdx.x;                // b*LEN + q
    const int h   = threadIdx.x >> 5;
    const int d   = threadIdx.x & 31;
    const int b   = tok / LEN_TOK;

    const int Hs[4] = {100, 50, 25, 13};
    const int Ws[4] = {100, 50, 25, 13};
    const int Ss[4] = {0, 10000, 12500, 13125};

    const float* vbase = value + (size_t)b * LEN_TOK * EMB;
    const int ch = h * HEAD_DIM + d;
    float acc = 0.f;

#pragma unroll
    for (int l = 0; l < 4; ++l) {
        const int W_ = Ws[l], H_ = Hs[l], s = Ss[l];
        const float rx = refp[((size_t)tok * N_LVLS + l) * 2 + 0];
        const float ry = refp[((size_t)tok * N_LVLS + l) * 2 + 1];
#pragma unroll
        for (int p = 0; p < 4; ++p) {
            size_t oi = (size_t)tok * 256 + (size_t)((h * N_LVLS + l) * N_POINTS + p) * 2;
            float ox = off[oi], oy = off[oi + 1];
            float a  = attn[(size_t)tok * 128 + h * 16 + l * 4 + p];

            float x = (rx + ox / (float)W_) * (float)W_ - 0.5f;
            float y = (ry + oy / (float)H_) * (float)H_ - 0.5f;
            float x0f = floorf(x), y0f = floorf(y);
            float wx = x - x0f, wy = y - y0f;
            int x0 = (int)x0f, y0 = (int)y0f;

            float samp = 0.f;
#pragma unroll
            for (int cy = 0; cy < 2; ++cy) {
#pragma unroll
                for (int cx = 0; cx < 2; ++cx) {
                    int xi = x0 + cx, yi = y0 + cy;
                    float w = (cx ? wx : (1.f - wx)) * (cy ? wy : (1.f - wy));
                    bool valid = (xi >= 0) & (xi < W_) & (yi >= 0) & (yi < H_);
                    int xc = xi < 0 ? 0 : (xi > W_ - 1 ? W_ - 1 : xi);
                    int yc = yi < 0 ? 0 : (yi > H_ - 1 ? H_ - 1 : yi);
                    float g = vbase[(size_t)(s + yc * W_ + xc) * EMB + ch];
                    samp += g * (valid ? w : 0.f);
                }
            }
            acc += a * samp;
        }
    }
    samp_out[(size_t)tok * EMB + ch] = (__bf16)acc;
}

// =====================================================================
// Fused residual + LayerNorm. One wave per row (256 cols, 8 per lane).
// Optionally emits a bf16 copy for the next GEMM.
// =====================================================================
__global__ void add_ln_kernel(const float* __restrict__ a, const float* __restrict__ r,
                              const float* __restrict__ gam, const float* __restrict__ bet,
                              float* __restrict__ out_f32, __bf16* __restrict__ out_bf16,
                              int M) {
    int row = blockIdx.x * 8 + (threadIdx.x >> 5);
    if (row >= M) return;
    int lane = threadIdx.x & 31;
    const float* pa = a + (size_t)row * EMB;
    const float* pr = r + (size_t)row * EMB;

    float v[8];
    float s = 0.f;
#pragma unroll
    for (int i = 0; i < 8; ++i) {
        int c = lane + i * 32;
        v[i] = pa[c] + pr[c];
        s += v[i];
    }
#pragma unroll
    for (int o = 16; o > 0; o >>= 1) s += __shfl_xor(s, o, 32);
    float mu = s * (1.f / 256.f);

    float vs = 0.f;
#pragma unroll
    for (int i = 0; i < 8; ++i) { float d = v[i] - mu; vs += d * d; }
#pragma unroll
    for (int o = 16; o > 0; o >>= 1) vs += __shfl_xor(vs, o, 32);
    float rstd = rsqrtf(vs * (1.f / 256.f) + 1e-5f);

#pragma unroll
    for (int i = 0; i < 8; ++i) {
        int c = lane + i * 32;
        float y = (v[i] - mu) * rstd * gam[c] + bet[c];
        if (out_f32)  out_f32[(size_t)row * EMB + c] = y;
        if (out_bf16) out_bf16[(size_t)row * EMB + c] = (__bf16)y;
    }
}

// =====================================================================
// Host orchestration
// =====================================================================
static inline size_t align256(size_t x) { return (x + 255) & ~(size_t)255; }

extern "C" void kernel_launch(void* const* d_in, const int* in_sizes, int n_in,
                              void* d_out, int out_size, void* d_ws, size_t ws_size,
                              hipStream_t stream) {
    (void)in_sizes; (void)n_in; (void)out_size; (void)ws_size;
    const float* features = (const float*)d_in[0];
    const float* pos      = (const float*)d_in[1];
    const float* refp     = (const float*)d_in[2];
    // d_in[3] spatial_shapes, d_in[4] lvl_start_idx: compile-time constants here
    const float* W_off  = (const float*)d_in[5];
    const float* b_off  = (const float*)d_in[6];
    const float* W_attn = (const float*)d_in[7];
    const float* b_attn = (const float*)d_in[8];
    const float* W_val  = (const float*)d_in[9];
    const float* b_val  = (const float*)d_in[10];
    const float* W_out  = (const float*)d_in[11];
    const float* b_out  = (const float*)d_in[12];
    const float* ln1_g  = (const float*)d_in[13];
    const float* ln1_b  = (const float*)d_in[14];
    const float* W1     = (const float*)d_in[15];
    const float* b1     = (const float*)d_in[16];
    const float* W2     = (const float*)d_in[17];
    const float* b2     = (const float*)d_in[18];
    const float* ln2_g  = (const float*)d_in[19];
    const float* ln2_b  = (const float*)d_in[20];

    const size_t M = NTOK;
    char* ws = (char*)d_ws;
    size_t o = 0;
    const size_t szF32  = (size_t)M * EMB * sizeof(float);     // 54.5 MB
    const size_t szAttn = (size_t)M * 128 * sizeof(float);     // 27.2 MB
    const size_t szBF   = (size_t)M * EMB * sizeof(__bf16);    // 27.2 MB
    const size_t szH    = (size_t)M * D_FFN * sizeof(__bf16);  // 109 MB

    float*  buf0 = (float*)(ws + o);  o += align256(szF32);   // value -> attn_out -> ffn
    float*  buf1 = (float*)(ws + o);  o += align256(szF32);   // off   -> x1_f32
    float*  attn = (float*)(ws + o);  o += align256(szAttn);  // attn logits/softmax
    __bf16* buf3 = (__bf16*)(ws + o); o += align256(szBF);    // feat_bf16 -> samp_bf16 -> x1_bf16
    __bf16* q_bf = (__bf16*)(ws + o); o += align256(szBF);
    __bf16* h_bf = (__bf16*)(ws + o); o += align256(szH);
    __bf16* pWval  = (__bf16*)(ws + o); o += align256((size_t)256 * 256 * 2);
    __bf16* pWoff  = (__bf16*)(ws + o); o += align256((size_t)256 * 256 * 2);
    __bf16* pWattn = (__bf16*)(ws + o); o += align256((size_t)256 * 128 * 2);
    __bf16* pWout  = (__bf16*)(ws + o); o += align256((size_t)256 * 256 * 2);
    __bf16* pW1    = (__bf16*)(ws + o); o += align256((size_t)256 * 1024 * 2);
    __bf16* pW2    = (__bf16*)(ws + o); o += align256((size_t)1024 * 256 * 2);

    // ---- 1. pack weights into WMMA-B fragment order ----
    auto packGrid = [](int K, int N) { return dim3((unsigned)(((size_t)K * N + 255) / 256)); };
    pack_b_kernel<<<packGrid(256, 256),  256, 0, stream>>>(W_val,  pWval,  256, 256);
    pack_b_kernel<<<packGrid(256, 256),  256, 0, stream>>>(W_off,  pWoff,  256, 256);
    pack_b_kernel<<<packGrid(256, 128),  256, 0, stream>>>(W_attn, pWattn, 256, 128);
    pack_b_kernel<<<packGrid(256, 256),  256, 0, stream>>>(W_out,  pWout,  256, 256);
    pack_b_kernel<<<packGrid(256, 1024), 256, 0, stream>>>(W1,     pW1,    256, 1024);
    pack_b_kernel<<<packGrid(1024, 256), 256, 0, stream>>>(W2,     pW2,    1024, 256);

    // ---- 2. bf16 activations ----
    size_t nElem = M * EMB;
    prep_kernel<<<dim3((unsigned)((nElem + 255) / 256)), 256, 0, stream>>>(
        features, pos, buf3, q_bf, nElem);

    dim3 blk(256);
    unsigned gmx = (unsigned)((M + 127) / 128);

    // ---- 3-5. projection GEMMs (K=256, A read exactly once each) ----
    gemm_wmma_kc_kernel<2, false><<<dim3(gmx, 1), blk, 0, stream>>>(buf3, pWval,  b_val,  buf0, (int)M, 256); // value
    gemm_wmma_kc_kernel<2, false><<<dim3(gmx, 1), blk, 0, stream>>>(q_bf, pWoff,  b_off,  buf1, (int)M, 256); // offsets
    gemm_wmma_kc_kernel<1, false><<<dim3(gmx, 1), blk, 0, stream>>>(q_bf, pWattn, b_attn, attn, (int)M, 128); // attn logits

    // ---- 6. softmax over 16 (levels x points) per head ----
    int ngroups = (int)(M * N_HEADS);
    softmax16_kernel<<<dim3((unsigned)((ngroups + 255) / 256)), 256, 0, stream>>>(attn, ngroups);

    // ---- 7. deformable bilinear sampling (writes bf16) ----
    deform_sample_kernel<<<dim3((unsigned)M), 256, 0, stream>>>(buf0, buf1, attn, refp, buf3);

    // ---- 8. output projection ----
    gemm_wmma_kc_kernel<2, false><<<dim3(gmx, 1), blk, 0, stream>>>(buf3, pWout, b_out, buf0, (int)M, 256); // attn_out

    // ---- 9. residual + LN1 (fp32 residual + bf16 for FFN) ----
    add_ln_kernel<<<dim3((unsigned)((M + 7) / 8)), 256, 0, stream>>>(
        features, buf0, ln1_g, ln1_b, buf1, buf3, (int)M);

    // ---- 10-11. FFN ----
    gemm_wmma_kc_kernel<8, true><<<dim3(gmx, 1), blk, 0, stream>>>(buf3, pW1, b1, h_bf, (int)M, 1024);          // relu->bf16
    gemm_wmma_stream_kernel<false><<<dim3(gmx, 2), blk, 0, stream>>>(h_bf, pW2, b2, buf0, (int)M, 256, 1024);

    // ---- 12. residual + LN2 -> d_out ----
    add_ln_kernel<<<dim3((unsigned)((M + 7) / 8)), 256, 0, stream>>>(
        buf1, buf0, ln2_g, ln2_b, (float*)d_out, nullptr, (int)M);
}
